// VALSTM_56521769615653
// MI455X (gfx1250) — matrix operand order repaired and checked by reading, not compiled, not run on hardware
//
#include <hip/hip_runtime.h>
#include <hip/hip_bf16.h>
#include <cstdint>

typedef _Float16 v8h  __attribute__((ext_vector_type(8)));
typedef _Float16 v16h __attribute__((ext_vector_type(16)));
typedef float    v8f  __attribute__((ext_vector_type(8)));
typedef unsigned int u32x4 __attribute__((ext_vector_type(4)));
typedef int          i32x4 __attribute__((ext_vector_type(4)));
typedef int          i32x8 __attribute__((ext_vector_type(8)));

#define IN      64
#define HS      128
#define G4      512     // 4*HS
#define HC2     256     // 2*HS
#define BATCH   256
#define TSEQ    512
#define ROWS    16      // batch rows per workgroup (WMMA M)
#define NTHREADS 256    // 8 waves of 32
#define NBLOCKS (BATCH / ROWS)

#if defined(__has_builtin)
# if __has_builtin(__builtin_amdgcn_tensor_load_to_lds) && \
     __has_builtin(__builtin_amdgcn_s_wait_tensorcnt)
#  define HAVE_TDM 1
# endif
#endif
#ifndef HAVE_TDM
# define HAVE_TDM 0
#endif

// ---------------- dynamic LDS layout (byte offsets, all 16B aligned) ----------------
#define OFF_FRAGW   0                         //  64x512 f16 -> 65536 B
#define OFF_FRAGU   (OFF_FRAGW  + 65536)      // 128x512 f16 -> 131072 B
#define OFF_FRAGWA  (OFF_FRAGU  + 131072)     //  64x64  f16 -> 8192 B
#define OFF_FRAGUA  (OFF_FRAGWA + 8192)       // 256x64  f16 -> 32768 B
#define OFF_FRAGVA  (OFF_FRAGUA + 32768)      //  64x64  f16 -> 8192 B
#define OFF_BIAS    (OFF_FRAGVA + 8192)       // 512 f32 -> 2048 B
#define OFF_BA      (OFF_BIAS   + 2048)       // 64 f32  -> 256 B
#define OFF_X16     (OFF_BA     + 256)        // 16x64 f16 -> 2048 B
#define OFF_XG16    (OFF_X16    + 2048)       // 16x64 f16 -> 2048 B
#define OFF_XF0     (OFF_XG16   + 2048)       // 16x64 f32 stage buf 0 -> 4096 B
#define OFF_XF1     (OFF_XF0    + 4096)       // 16x64 f32 stage buf 1 -> 4096 B
#define OFF_HC16    (OFF_XF1    + 4096)       // 16x256 f16 (h|c) -> 8192 B
#define OFF_AACT    (OFF_HC16   + 8192)       // 16x64 f16 -> 2048 B
#define OFF_ARAW    (OFF_AACT   + 2048)       // 16x64 f32 -> 4096 B
#define OFF_GBUF    (OFF_ARAW   + 4096)       // 16x512 f32 -> 32768 B
#define OFF_CST     (OFF_GBUF   + 32768)      // 16x128 f32 -> 8192 B
#define OFF_HST     (OFF_CST    + 8192)       // 16x128 f32 -> 8192 B
#define SMEM_BYTES  (OFF_HST    + 8192)       // = 323840 <= 327680 (320KB/WGP)

static_assert(SMEM_BYTES <= 320 * 1024, "LDS overflow");

// A operand (16-bit, 16x32): lane L holds row M=L&15; halves j=0..7 are
// K = kb..kb+7, halves j=8..15 are K = kb+16..kb+23, kb = (L>>4)*8.
__device__ __forceinline__ v16h load_A16(const _Float16* __restrict__ base,
                                         int kstride, int row, int k0, int lane) {
    const int kb = (lane >> 4) << 3;          // 0 or 8
    const _Float16* p = base + row * kstride + k0 + kb;
    v8h lo = *(const v8h*)p;
    v8h hi = *(const v8h*)(p + 16);
    return __builtin_shufflevector(lo, hi, 0,1,2,3,4,5,6,7,8,9,10,11,12,13,14,15);
}

// B operand stored pre-swizzled: tile stride = 512 halves, lane stride = 16 halves.
__device__ __forceinline__ v16h load_Bfrag(const _Float16* __restrict__ base,
                                           int tile, int lane) {
    const _Float16* p = base + (((size_t)tile << 5) + lane) * 16;
    v8h lo = *(const v8h*)p;
    v8h hi = *(const v8h*)(p + 8);
    return __builtin_shufflevector(lo, hi, 0,1,2,3,4,5,6,7,8,9,10,11,12,13,14,15);
}

__device__ __forceinline__ v8f wmma_f16(v16h a, v16h b, v8f c) {
    return __builtin_amdgcn_wmma_f32_16x16x32_f16(false, a, false, b,
                                                  (short)0, c, false, false);
}

// TDM: DMA a 16-row x 64-col f32 tile (row pitch TSEQ*IN elements) into LDS.
__device__ __forceinline__ void tdm_load_xtile(const float* gsrc, void* ldsdst) {
#if HAVE_TDM
    unsigned long long ga = (unsigned long long)(uintptr_t)gsrc;
    unsigned int ldsoff = (unsigned int)(uintptr_t)ldsdst; // addr[31:0] = LDS byte offset
    u32x4 g0;
    g0[0] = 1u;                                            // count=1 (valid descriptor)
    g0[1] = ldsoff;                                        // lds_addr
    g0[2] = (unsigned int)(ga & 0xFFFFFFFFu);              // global_addr[31:0]
    g0[3] = (unsigned int)((ga >> 32) & 0x1FFFFFFu)        // global_addr[56:32]
          | 0x80000000u;                                   // type=2 ("image")
    i32x8 g1;
    g1[0] = 0x00020000;                 // data_size=2 (4-byte elements)
    g1[1] = (int)(64u << 16);           // tensor_dim0 = 64  (bits 79:48)
    g1[2] = (int)(16u << 16);           // tensor_dim1 = 16  (bits 111:80)
    g1[3] = (int)(64u << 16);           // tile_dim0  = 64  (bits 127:112)
    g1[4] = 16;                         // tile_dim1  = 16  (bits 143:128)
    g1[5] = (int)(TSEQ * IN);           // tensor_dim0_stride = 32768 (bits 207:160)
    g1[6] = 0;
    g1[7] = 0;
    i32x4 z4 = {0, 0, 0, 0};            // VADDR2/VADDR3 unused (2D tensor)
    i32x8 z8 = {0, 0, 0, 0, 0, 0, 0, 0};
    __builtin_amdgcn_tensor_load_to_lds(g0, g1, z4, z4, z8, 0);
#else
    (void)gsrc; (void)ldsdst;
#endif
}

// Pack a row-major f32 [Ktot][Ntot] weight into f16 B-fragment order.
__device__ void pack_weight(const float* __restrict__ src, int Ktot, int Ntot,
                            _Float16* __restrict__ dst, int tid) {
    const int numNT = Ntot >> 4;
    const int total = Ktot * Ntot;
    for (int idx = tid; idx < total; idx += NTHREADS) {
        int tile = idx >> 9;
        int rem  = idx & 511;
        int lane = rem >> 4;
        int j    = rem & 15;
        int kt = tile / numNT;
        int nt = tile - kt * numNT;
        int krow = (kt << 5) + ((lane >> 4) << 3) + (j & 7) + ((j >> 3) << 4);
        int ncol = (nt << 4) + (lane & 15);
        dst[idx] = (_Float16)src[(size_t)krow * Ntot + ncol];
    }
}

__device__ __forceinline__ float sigmoidf_fast(float x) {
    return 1.0f / (1.0f + __expf(-x));
}

__global__ __launch_bounds__(NTHREADS)
void VALSTM_56521769615653_kernel(const float* __restrict__ X,
                                  const float* __restrict__ W,
                                  const float* __restrict__ U,
                                  const float* __restrict__ bias,
                                  const float* __restrict__ Wa,
                                  const float* __restrict__ Ua,
                                  const float* __restrict__ ba,
                                  const float* __restrict__ Va,
                                  const float* __restrict__ fc_w,
                                  const float* __restrict__ fc_b,
                                  float* __restrict__ out_y,
                                  float* __restrict__ out_hidden,
                                  float* __restrict__ out_ht,
                                  float* __restrict__ out_ct,
                                  float* __restrict__ out_alpha) {
    extern __shared__ char smem[];
    _Float16* fragW  = (_Float16*)(smem + OFF_FRAGW);
    _Float16* fragU  = (_Float16*)(smem + OFF_FRAGU);
    _Float16* fragWa = (_Float16*)(smem + OFF_FRAGWA);
    _Float16* fragUa = (_Float16*)(smem + OFF_FRAGUA);
    _Float16* fragVa = (_Float16*)(smem + OFF_FRAGVA);
    float*    bias_l = (float*)   (smem + OFF_BIAS);
    float*    ba_l   = (float*)   (smem + OFF_BA);
    _Float16* x16    = (_Float16*)(smem + OFF_X16);
    _Float16* xg16   = (_Float16*)(smem + OFF_XG16);
    float*    xfbuf0 = (float*)   (smem + OFF_XF0);
    float*    xfbuf1 = (float*)   (smem + OFF_XF1);
    _Float16* hc16   = (_Float16*)(smem + OFF_HC16);   // [16][256]: h | c
    _Float16* aact16 = (_Float16*)(smem + OFF_AACT);
    float*    araw   = (float*)   (smem + OFF_ARAW);
    float*    gbuf   = (float*)   (smem + OFF_GBUF);
    float*    cst    = (float*)   (smem + OFF_CST);
    float*    hst    = (float*)   (smem + OFF_HST);

    const int tid  = threadIdx.x;
    const int lane = tid & 31;
    const int wv   = tid >> 5;
    const int rowbase = blockIdx.x * ROWS;

    // ---- one-time: pack weights into LDS f16 fragments, load biases, zero state
    pack_weight(W,  IN,  G4, fragW,  tid);   // numNT=32, numKT=2
    pack_weight(U,  HS,  G4, fragU,  tid);   // numNT=32, numKT=4
    pack_weight(Wa, IN,  IN, fragWa, tid);   // numNT=4,  numKT=2
    pack_weight(Ua, HC2, IN, fragUa, tid);   // numNT=4,  numKT=8
    pack_weight(Va, IN,  IN, fragVa, tid);   // numNT=4,  numKT=2
    for (int i = tid; i < G4; i += NTHREADS) bias_l[i] = bias[i];
    for (int i = tid; i < IN; i += NTHREADS) ba_l[i] = ba[i];
    for (int i = tid; i < ROWS * HS; i += NTHREADS) { cst[i] = 0.0f; hst[i] = 0.0f; }
    for (int i = tid; i < ROWS * HC2; i += NTHREADS) hc16[i] = (_Float16)0.0f;

#if HAVE_TDM
    if (wv == 0)   // prime the pipeline: DMA x_0 into stage buffer 0
        tdm_load_xtile(X + (size_t)rowbase * TSEQ * IN, xfbuf0);
#endif
    __syncthreads();

    const int arow  = lane & 15;              // A-operand row for this lane
    const int mrow0 = (lane >> 4) << 3;       // C-fragment M base (0 or 8)
    const int ncolL = lane & 15;              // C-fragment N within tile

    for (int t = 0; t < TSEQ; ++t) {
        float* xff = (t & 1) ? xfbuf1 : xfbuf0;

#if HAVE_TDM
        // ---- phase 1: wait for x_t DMA, kick off x_{t+1} DMA, convert to f16
        if (wv == 0) {
            __builtin_amdgcn_s_wait_tensorcnt(0);
            if (t + 1 < TSEQ)
                tdm_load_xtile(X + ((size_t)rowbase * TSEQ + (t + 1)) * IN,
                               (t & 1) ? (void*)xfbuf0 : (void*)xfbuf1);
        }
        __syncthreads();   // publish TDM-written LDS to all waves
        for (int e = tid; e < ROWS * IN; e += NTHREADS)
            x16[e] = (_Float16)xff[e];
#else
        // ---- phase 1 (fallback): stage x_t via vector loads
        for (int e = tid; e < ROWS * IN; e += NTHREADS) {
            int r = e >> 6, col = e & 63;
            size_t gidx = ((size_t)(rowbase + r) * TSEQ + t) * IN + col;
            float v = X[gidx];
            xff[e] = v;
            x16[e] = (_Float16)v;
            if (t + 1 < TSEQ) __builtin_prefetch(&X[gidx + IN], 0, 0);
        }
#endif
        __syncthreads();

        // ---- phase 2: apre = tanh(x@Wa + [h,c]@Ua + ba)   (waves 0..3)
        if (wv < 4) {
            const int nt = wv, n0 = nt << 4;
            v8f acc = {};
            #pragma unroll
            for (int kt = 0; kt < 2; ++kt)
                acc = wmma_f16(load_A16(x16, IN, arow, kt << 5, lane),
                               load_Bfrag(fragWa, kt * 4 + nt, lane), acc);
            #pragma unroll
            for (int kt = 0; kt < 8; ++kt)
                acc = wmma_f16(load_A16(hc16, HC2, arow, kt << 5, lane),
                               load_Bfrag(fragUa, kt * 4 + nt, lane), acc);
            float bv = ba_l[n0 + ncolL];
            #pragma unroll
            for (int r = 0; r < 8; ++r)
                aact16[(mrow0 + r) * IN + n0 + ncolL] = (_Float16)tanhf(acc[r] + bv);
        }
        __syncthreads();

        // ---- phase 3: a_t = apre @ Va
        if (wv < 4) {
            const int nt = wv, n0 = nt << 4;
            v8f acc = {};
            #pragma unroll
            for (int kt = 0; kt < 2; ++kt)
                acc = wmma_f16(load_A16(aact16, IN, arow, kt << 5, lane),
                               load_Bfrag(fragVa, kt * 4 + nt, lane), acc);
            #pragma unroll
            for (int r = 0; r < 8; ++r)
                araw[(mrow0 + r) * IN + n0 + ncolL] = acc[r];
        }
        __syncthreads();

        // ---- phase 4: parallel softmax (16 half-wave groups, one row each),
        //      xg = alpha * x_t
        {
            const int g   = tid >> 4;     // row 0..15
            const int sub = tid & 15;     // lane within 16-lane group
            float v[4];
            #pragma unroll
            for (int k = 0; k < 4; ++k) v[k] = araw[g * IN + sub + (k << 4)];
            float m = fmaxf(fmaxf(v[0], v[1]), fmaxf(v[2], v[3]));
            #pragma unroll
            for (int d = 1; d < 16; d <<= 1) m = fmaxf(m, __shfl_xor(m, d, 32));
            float e[4], s = 0.0f;
            #pragma unroll
            for (int k = 0; k < 4; ++k) { e[k] = __expf(v[k] - m); s += e[k]; }
            #pragma unroll
            for (int d = 1; d < 16; d <<= 1) s += __shfl_xor(s, d, 32);
            float inv = 1.0f / s;
            #pragma unroll
            for (int k = 0; k < 4; ++k) {
                int idx = g * IN + sub + (k << 4);
                float al = e[k] * inv;
                xg16[idx] = (_Float16)(al * xff[idx]);
                if (t == TSEQ - 1)
                    out_alpha[(size_t)(rowbase + g) * IN + sub + (k << 4)] = al;
            }
        }
        __syncthreads();

        // ---- phase 5: gates = xg@W + h@U + bias   (8 waves x 4 N-tiles)
        #pragma unroll
        for (int nn = 0; nn < 4; ++nn) {
            const int nt = wv + (nn << 3);
            const int n0 = nt << 4;
            v8f acc = {};
            #pragma unroll
            for (int kt = 0; kt < 2; ++kt)
                acc = wmma_f16(load_A16(xg16, IN, arow, kt << 5, lane),
                               load_Bfrag(fragW, (kt << 5) + nt, lane), acc);
            #pragma unroll
            for (int kt = 0; kt < 4; ++kt)   // h lives in hc16[:, 0:128]
                acc = wmma_f16(load_A16(hc16, HC2, arow, kt << 5, lane),
                               load_Bfrag(fragU, (kt << 5) + nt, lane), acc);
            float bv = bias_l[n0 + ncolL];
            #pragma unroll
            for (int r = 0; r < 8; ++r)
                gbuf[(mrow0 + r) * G4 + n0 + ncolL] = acc[r] + bv;
        }
        __syncthreads();

        // ---- phase 6: LSTM elementwise update (f32 state), emit hidden_seq
        for (int e = tid; e < ROWS * HS; e += NTHREADS) {
            int r = e >> 7, hcol = e & 127;
            float ig = sigmoidf_fast(gbuf[r * G4 + hcol]);
            float fg = sigmoidf_fast(gbuf[r * G4 + HS + hcol]);
            float gg = tanhf(gbuf[r * G4 + 2 * HS + hcol]);
            float og = sigmoidf_fast(gbuf[r * G4 + 3 * HS + hcol]);
            float c = fg * cst[e] + ig * gg;
            float h = og * tanhf(c);
            cst[e] = c;
            hst[e] = h;
            hc16[r * HC2 + hcol]      = (_Float16)h;
            hc16[r * HC2 + HS + hcol] = (_Float16)c;
            out_hidden[((size_t)(rowbase + r) * TSEQ + t) * HS + hcol] = h;
        }
        __syncthreads();
    }

    // ---- epilogue: h_T, c_T, y = h_T @ fc_w + fc_b
    for (int e = tid; e < ROWS * HS; e += NTHREADS) {
        int r = e >> 7, hcol = e & 127;
        out_ht[(size_t)(rowbase + r) * HS + hcol] = hst[e];
        out_ct[(size_t)(rowbase + r) * HS + hcol] = cst[e];
    }
    if (tid < ROWS) {
        const int r = tid;
        float acc = fc_b[0];
        for (int k = 0; k < HS; ++k) acc += hst[r * HS + k] * fc_w[k];
        out_y[rowbase + r] = acc;
    }
}

extern "C" void kernel_launch(void* const* d_in, const int* in_sizes, int n_in,
                              void* d_out, int out_size, void* d_ws, size_t ws_size,
                              hipStream_t stream) {
    (void)in_sizes; (void)n_in; (void)out_size; (void)d_ws; (void)ws_size;
    const float* X    = (const float*)d_in[0];
    const float* W    = (const float*)d_in[1];
    const float* U    = (const float*)d_in[2];
    const float* bias = (const float*)d_in[3];
    const float* Wa   = (const float*)d_in[4];
    const float* Ua   = (const float*)d_in[5];
    const float* ba   = (const float*)d_in[6];
    const float* Va   = (const float*)d_in[7];
    const float* fc_w = (const float*)d_in[8];
    const float* fc_b = (const float*)d_in[9];

    float* out        = (float*)d_out;
    float* out_y      = out;                                    // [256]
    float* out_hidden = out_y + BATCH;                          // [256*512*128]
    float* out_ht     = out_hidden + (size_t)BATCH * TSEQ * HS; // [256*128]
    float* out_ct     = out_ht + (size_t)BATCH * HS;            // [256*128]
    float* out_alpha  = out_ct + (size_t)BATCH * HS;            // [256*64]

    (void)hipFuncSetAttribute((const void*)VALSTM_56521769615653_kernel,
                              hipFuncAttributeMaxDynamicSharedMemorySize, SMEM_BYTES);

    VALSTM_56521769615653_kernel<<<NBLOCKS, NTHREADS, SMEM_BYTES, stream>>>(
        X, W, U, bias, Wa, Ua, ba, Va, fc_w, fc_b,
        out_y, out_hidden, out_ht, out_ct, out_alpha);
}
